// MultiHeadAttention_6279242187389
// MI455X (gfx1250) — compile-verified, matrix operand order checked by reading, not run on hardware
//
#include <hip/hip_runtime.h>

// ---------------------------------------------------------------------------
// Problem constants (match reference)
// ---------------------------------------------------------------------------
constexpr int CB  = 4;      // batch
constexpr int CS  = 2048;   // sequence
constexpr int CD  = 1024;   // d_model
constexpr int CH  = 16;     // heads
constexpr int CDH = 64;     // head dim
constexpr int CM  = CB * CS;        // 8192 rows for the projections

// ---------------------------------------------------------------------------
// Types
// ---------------------------------------------------------------------------
typedef __bf16 bf16_t;
typedef __attribute__((ext_vector_type(8)))  __bf16 v8bf;
typedef __attribute__((ext_vector_type(16))) __bf16 v16bf;
typedef __attribute__((ext_vector_type(8)))  float  v8f;
typedef __attribute__((ext_vector_type(4)))  int    v4i;

// CDNA5 async global->LDS copy path (ASYNCcnt), guarded so the file compiles
// even if this toolchain doesn't declare the builtins.
#if defined(__gfx1250__) && __has_builtin(__builtin_amdgcn_global_load_async_to_lds_b128) && __has_builtin(__builtin_amdgcn_s_wait_asynccnt)
#define USE_ASYNC_LDS 1
#else
#define USE_ASYNC_LDS 0
#endif

static __device__ inline bf16_t f2bf(float f) {
    union { float f; unsigned u; } a;
    a.f = f;
    unsigned r = (a.u + 0x7FFFu + ((a.u >> 16) & 1u)) >> 16;  // RNE
    union { unsigned short s; bf16_t b; } o;
    o.s = (unsigned short)r;
    return o.b;
}

static __device__ inline v16bf frag_combine(v8bf lo, v8bf hi) {
    return __builtin_shufflevector(lo, hi, 0,1,2,3,4,5,6,7,8,9,10,11,12,13,14,15);
}

// Load a 16x32 bf16 WMMA A/B fragment from a row-major matrix (global or LDS).
// Lane layout per ISA 7.12.2: lanes 0-15 hold K {k0..k0+7, k0+16..k0+23},
// lanes 16-31 hold K {k0+8..k0+15, k0+24..k0+31}.
static __device__ inline v16bf load_frag(const bf16_t* p, int ld, int r0, int k0) {
    const int lane = threadIdx.x & 31;
    const int hl   = (lane >> 4) & 1;
    const int r    = r0 + (lane & 15);
    const bf16_t* base = p + (size_t)r * ld + k0 + hl * 8;
    v8bf lo = *(const v8bf*)(base);
    v8bf hi = *(const v8bf*)(base + 16);
    return frag_combine(lo, hi);
}

static __device__ inline v8f wmma_bf16(v16bf a, v16bf b, v8f c) {
    return __builtin_amdgcn_wmma_f32_16x16x32_bf16(
        /*neg_a=*/false, a, /*neg_b=*/false, b,
        /*c_mod=*/(short)0, c, /*reuse_a=*/false, /*reuse_b=*/false);
}

// 16-byte global->LDS copy: async on CDNA5 (ASYNCcnt), sync fallback otherwise.
static __device__ inline void copy16_g2l(const bf16_t* g, bf16_t* l) {
#if USE_ASYNC_LDS
    __builtin_amdgcn_global_load_async_to_lds_b128(
        (__attribute__((address_space(1))) v4i*)(g),
        (__attribute__((address_space(3))) v4i*)(l),
        0, 0);
#else
    *(v8bf*)l = *(const v8bf*)g;
#endif
}

static __device__ inline void async_wait0() {
#if USE_ASYNC_LDS
    __builtin_amdgcn_s_wait_asynccnt(0);
#endif
}

// ---------------------------------------------------------------------------
// Stage 0: fp32 -> bf16 conversion
// ---------------------------------------------------------------------------
__global__ void cvt_f32_to_bf16(const float* __restrict__ src, bf16_t* __restrict__ dst, int n) {
    int i = blockIdx.x * blockDim.x + threadIdx.x;
    if (i < n) dst[i] = f2bf(src[i]);
}

// ---------------------------------------------------------------------------
// Stage 1/5: C[M,N] (f32) = A[M,K] (bf16) * B[N,K]^T (bf16)   (nn.Linear)
// Block: 256 threads = 8 waves, block tile 128x128, wave tile 32x64.
// A/B K-slices (128x32) staged into double-buffered LDS via async copies;
// WMMA fragments are fed by ds_load_b128 out of LDS.
// ---------------------------------------------------------------------------
constexpr int GK  = 32;   // K per pipeline stage (one WMMA k-step)
constexpr int LDT = 40;   // padded LDS row stride (elems): 80B, 16B-aligned

__global__ __launch_bounds__(256) void gemm_bf16_nt(
    const bf16_t* __restrict__ A, const bf16_t* __restrict__ Bw,
    float* __restrict__ C, int M, int N, int K) {
    __shared__ __align__(16) bf16_t lA[2][128 * LDT];
    __shared__ __align__(16) bf16_t lB[2][128 * LDT];

    const int tid  = threadIdx.x;
    const int lane = tid & 31;
    const int hl   = (lane >> 4) & 1;
    const int wave = tid >> 5;
    const int wm   = wave & 3;               // 4 waves along M
    const int wn   = wave >> 2;              // 2 waves along N
    const int bm   = blockIdx.x * 128;
    const int bn   = blockIdx.y * 128;

    // Stage one 128x32 slice of A and B into lA/lB[buf].
    // 128 rows x 4 chunks of 8 bf16 = 512 16B-chunks per tile; 256 threads.
    auto stage = [&](int buf, int k0) {
#pragma unroll
        for (int c = 0; c < 2; ++c) {
            const int idx = tid + c * 256;       // 0..511
            const int row = idx >> 2;            // 0..127
            const int ch  = (idx & 3) * 8;       // 0,8,16,24
            copy16_g2l(A  + (size_t)(bm + row) * K + k0 + ch, &lA[buf][row * LDT + ch]);
            copy16_g2l(Bw + (size_t)(bn + row) * K + k0 + ch, &lB[buf][row * LDT + ch]);
        }
    };

    v8f acc[2][4];
#pragma unroll
    for (int mi = 0; mi < 2; ++mi)
#pragma unroll
        for (int j = 0; j < 4; ++j)
#pragma unroll
            for (int e = 0; e < 8; ++e) acc[mi][j][e] = 0.0f;

    stage(0, 0);
    int buf = 0;
    for (int k0 = 0; k0 < K; k0 += GK) {
        async_wait0();          // our async copies into lA/lB[buf] have landed
        __syncthreads();        // all threads' copies visible; prev reads done

        if (k0 + GK < K) {
            stage(buf ^ 1, k0 + GK);
            // pull the stage after next toward L2 (speculative prefetch)
            if (k0 + 2 * GK < K) {
                __builtin_prefetch(A  + (size_t)(bm + (tid >> 1)) * K + k0 + 2 * GK, 0, 1);
                __builtin_prefetch(Bw + (size_t)(bn + (tid >> 1)) * K + k0 + 2 * GK, 0, 1);
            }
        }

        // Compute on lA/lB[buf]: wave tile 32x64 = 2x4 WMMA tiles, k-step 32.
        v16bf a0 = load_frag(&lA[buf][0], LDT, wm * 32,      0);
        v16bf a1 = load_frag(&lA[buf][0], LDT, wm * 32 + 16, 0);
#pragma unroll
        for (int j = 0; j < 4; ++j) {
            v16bf b = load_frag(&lB[buf][0], LDT, wn * 64 + j * 16, 0);
            acc[0][j] = wmma_bf16(a0, b, acc[0][j]);
            acc[1][j] = wmma_bf16(a1, b, acc[1][j]);
        }
        buf ^= 1;
    }

#pragma unroll
    for (int mi = 0; mi < 2; ++mi)
#pragma unroll
        for (int j = 0; j < 4; ++j) {
            const int row0 = bm + wm * 32 + mi * 16 + hl * 8;
            const int col  = bn + wn * 64 + j * 16 + (lane & 15);
#pragma unroll
            for (int i = 0; i < 8; ++i)
                C[(size_t)(row0 + i) * N + col] = acc[mi][j][i];
        }
}

// ---------------------------------------------------------------------------
// Stage 2: RoPE on q,k + layout change; v transposed for the PV B-fragment.
//   qf/kf/vf: [B,S,D] f32   ->  qb/kb: [B,H,S,Dh] bf16, vtb: [B,H,Dh,S] bf16
// ---------------------------------------------------------------------------
__global__ void rope_reshape(const float* __restrict__ qf, const float* __restrict__ kf,
                             const float* __restrict__ vf,
                             const float* __restrict__ cosp, const float* __restrict__ sinp,
                             bf16_t* __restrict__ qb, bf16_t* __restrict__ kb,
                             bf16_t* __restrict__ vtb) {
    int idx = blockIdx.x * blockDim.x + threadIdx.x;
    constexpr int total = CB * CS * CD;
    if (idx >= total) return;
    const int e  = idx & (CD - 1);
    const int s  = (idx / CD) & (CS - 1);
    const int b  = idx / (CD * CS);
    const int h  = e >> 6;
    const int dh = e & 63;

    const float c  = cosp[s * CDH + dh];
    const float sn = sinp[s * CDH + dh];
    const float qv = qf[idx], kv = kf[idx], vv = vf[idx];
    const float qr = (dh < 32) ? -qf[idx + 32] : qf[idx - 32];
    const float kr = (dh < 32) ? -kf[idx + 32] : kf[idx - 32];

    const size_t head = (size_t)(b * CH + h);
    qb[(head * CS + s) * CDH + dh] = f2bf(qv * c + qr * sn);
    kb[(head * CS + s) * CDH + dh] = f2bf(kv * c + kr * sn);
    vtb[(head * CDH + dh) * CS + s] = f2bf(vv);
}

// ---------------------------------------------------------------------------
// Stage 3: flash attention (causal). One wave = 16 queries x full Dh=64.
// Block = 8 waves = 128 queries of one (b,h). Online softmax; probabilities
// transposed C->A layout through a private LDS strip per wave.
// Output written bf16 to attnb [B,S,D] for the final projection GEMM.
// ---------------------------------------------------------------------------
__global__ __launch_bounds__(256) void attn_flash(
    const bf16_t* __restrict__ qb, const bf16_t* __restrict__ kb,
    const bf16_t* __restrict__ vtb, bf16_t* __restrict__ attnb) {
    // per-wave LDS strip: 16 rows x 32 probs, row stride 48 elems (96B, 16B aligned)
    __shared__ __align__(16) bf16_t lds_p[8 * 16 * 48];

    const int lane = threadIdx.x & 31;
    const int hl   = (lane >> 4) & 1;
    const int wave = threadIdx.x >> 5;

    const int bh = blockIdx.y;               // b*CH + h
    const int b  = bh >> 4;
    const int h  = bh & 15;
    const bf16_t* qh = qb  + (size_t)bh * CS * CDH;
    const bf16_t* kh = kb  + (size_t)bh * CS * CDH;
    const bf16_t* vh = vtb + (size_t)bh * CDH * CS;

    const int q0 = blockIdx.x * 128 + wave * 16;

    // Q fragments for the wave's 16 rows (k-steps dh 0..31, 32..63)
    v16bf aq0 = load_frag(qh, CDH, q0, 0);
    v16bf aq1 = load_frag(qh, CDH, q0, 32);

    v8f o[4];
    float mrow[8], lrow[8];
#pragma unroll
    for (int j = 0; j < 4; ++j)
#pragma unroll
        for (int e = 0; e < 8; ++e) o[j][e] = 0.0f;
#pragma unroll
    for (int i = 0; i < 8; ++i) { mrow[i] = -3.0e38f; lrow[i] = 0.0f; }

    bf16_t* pw = lds_p + wave * (16 * 48);
    const int kmax = q0 + 15;                // causal: highest needed key

    for (int kb0 = 0; kb0 <= kmax; kb0 += 32) {
        // ---- scores S = Q K^T for 16q x 32k (two 16-key tiles) ----
        v8f sc[2];
#pragma unroll
        for (int n = 0; n < 2; ++n) {
            v8f c;
#pragma unroll
            for (int e = 0; e < 8; ++e) c[e] = 0.0f;
            v16bf bk0 = load_frag(kh, CDH, kb0 + n * 16, 0);
            v16bf bk1 = load_frag(kh, CDH, kb0 + n * 16, 32);
            c = wmma_bf16(aq0, bk0, c);
            c = wmma_bf16(aq1, bk1, c);
            sc[n] = c;
        }

        // ---- online softmax update, write probs (bf16) to LDS ----
#pragma unroll
        for (int i = 0; i < 8; ++i) {
            const int r    = i + hl * 8;     // row within 16-query tile
            const int qrow = q0 + r;
            const int key0 = kb0 + (lane & 15);
            float s0 = sc[0][i] * 0.125f;    // 1/sqrt(64)
            float s1 = sc[1][i] * 0.125f;
            if (key0      > qrow) s0 = -1.0e30f;
            if (key0 + 16 > qrow) s1 = -1.0e30f;

            float t = fmaxf(s0, s1);
#pragma unroll
            for (int off = 1; off < 16; off <<= 1)
                t = fmaxf(t, __shfl_xor(t, off, 32));
            const float mnew = fmaxf(mrow[i], t);

            const float p0 = __expf(s0 - mnew);
            const float p1 = __expf(s1 - mnew);
            float sum = p0 + p1;
#pragma unroll
            for (int off = 1; off < 16; off <<= 1)
                sum += __shfl_xor(sum, off, 32);

            const float alpha = __expf(mrow[i] - mnew);
            lrow[i] = lrow[i] * alpha + sum;
            mrow[i] = mnew;
#pragma unroll
            for (int j = 0; j < 4; ++j) o[j][i] *= alpha;

            pw[r * 48 +      (lane & 15)] = f2bf(p0);
            pw[r * 48 + 16 + (lane & 15)] = f2bf(p1);
        }

        // ---- transpose: read P back as an A-layout fragment (16x32) ----
        const bf16_t* prow = pw + (lane & 15) * 48 + hl * 8;
        v8bf plo = *(const v8bf*)(prow);
        v8bf phi = *(const v8bf*)(prow + 16);
        v16bf ap = frag_combine(plo, phi);

        // ---- O += P V  (4 WMMAs: Dh = 4 x 16 columns) ----
#pragma unroll
        for (int j = 0; j < 4; ++j) {
            v16bf bv = load_frag(vh, CS, j * 16, kb0);  // rows=dh, contiguous K=keys
            o[j] = wmma_bf16(ap, bv, o[j]);
        }
    }

    // ---- normalize and store bf16 into [B,S,D] ----
#pragma unroll
    for (int j = 0; j < 4; ++j) {
        const int col = j * 16 + (lane & 15);
#pragma unroll
        for (int i = 0; i < 8; ++i) {
            const int r = i + hl * 8;
            const float val = o[j][i] / lrow[i];
            attnb[(size_t)(b * CS + q0 + r) * CD + h * CDH + col] = f2bf(val);
        }
    }
}

// ---------------------------------------------------------------------------
// Host side
// ---------------------------------------------------------------------------
extern "C" void kernel_launch(void* const* d_in, const int* in_sizes, int n_in,
                              void* d_out, int out_size, void* d_ws, size_t ws_size,
                              hipStream_t stream) {
    const float* x    = (const float*)d_in[0];
    // d_in[1] = mask (causality reproduced analytically: 0 / -1e9)
    const float* wq   = (const float*)d_in[2];
    const float* wk   = (const float*)d_in[3];
    const float* wv   = (const float*)d_in[4];
    const float* wo   = (const float*)d_in[5];
    const float* cosp = (const float*)d_in[6];
    const float* sinp = (const float*)d_in[7];
    float* out = (float*)d_out;

    char* ws = (char*)d_ws;
    size_t off = 0;
    auto alloc = [&](size_t bytes) -> void* {
        void* p = ws + off;
        off += (bytes + 255) & ~(size_t)255;
        return p;
    };

    bf16_t* xb   = (bf16_t*)alloc((size_t)CM * CD * sizeof(bf16_t));
    bf16_t* wqb  = (bf16_t*)alloc((size_t)CD * CD * sizeof(bf16_t));
    bf16_t* wkb  = (bf16_t*)alloc((size_t)CD * CD * sizeof(bf16_t));
    bf16_t* wvb  = (bf16_t*)alloc((size_t)CD * CD * sizeof(bf16_t));
    bf16_t* wob  = (bf16_t*)alloc((size_t)CD * CD * sizeof(bf16_t));
    float*  qf   = (float*) alloc((size_t)CM * CD * sizeof(float));
    float*  kf   = (float*) alloc((size_t)CM * CD * sizeof(float));
    float*  vf   = (float*) alloc((size_t)CM * CD * sizeof(float));
    bf16_t* qrb  = (bf16_t*)alloc((size_t)CM * CD * sizeof(bf16_t));  // [B,H,S,Dh]
    bf16_t* krb  = (bf16_t*)alloc((size_t)CM * CD * sizeof(bf16_t));  // [B,H,S,Dh]
    bf16_t* vtb  = (bf16_t*)alloc((size_t)CM * CD * sizeof(bf16_t));  // [B,H,Dh,S]
    bf16_t* attb = (bf16_t*)alloc((size_t)CM * CD * sizeof(bf16_t));  // [B,S,D]

    const int nx = CM * CD;        // 8388608
    const int nw = CD * CD;        // 1048576

    // Stage 0: conversions
    cvt_f32_to_bf16<<<(nx + 255) / 256, 256, 0, stream>>>(x,  xb,  nx);
    cvt_f32_to_bf16<<<(nw + 255) / 256, 256, 0, stream>>>(wq, wqb, nw);
    cvt_f32_to_bf16<<<(nw + 255) / 256, 256, 0, stream>>>(wk, wkb, nw);
    cvt_f32_to_bf16<<<(nw + 255) / 256, 256, 0, stream>>>(wv, wvb, nw);
    cvt_f32_to_bf16<<<(nw + 255) / 256, 256, 0, stream>>>(wo, wob, nw);

    // Stage 1: Q/K/V projections (X * W^T)
    dim3 ggrid(CM / 128, CD / 128);
    gemm_bf16_nt<<<ggrid, 256, 0, stream>>>(xb, wqb, qf, CM, CD, CD);
    gemm_bf16_nt<<<ggrid, 256, 0, stream>>>(xb, wkb, kf, CM, CD, CD);
    gemm_bf16_nt<<<ggrid, 256, 0, stream>>>(xb, wvb, vf, CM, CD, CD);

    // Stage 2: RoPE + reshape/transpose
    rope_reshape<<<(nx + 255) / 256, 256, 0, stream>>>(qf, kf, vf, cosp, sinp,
                                                       qrb, krb, vtb);

    // Stage 3: flash attention
    dim3 agrid(CS / 128, CB * CH);
    attn_flash<<<agrid, 256, 0, stream>>>(qrb, krb, vtb, attb);

    // Stage 4: output projection -> d_out (fp32)
    gemm_bf16_nt<<<ggrid, 256, 0, stream>>>(attb, wob, out, CM, CD, CD);

    (void)in_sizes; (void)n_in; (void)out_size; (void)ws_size;
}
